// LlamaAttention_76501957476360
// MI455X (gfx1250) — matrix operand order repaired and checked by reading, not compiled, not run on hardware
//
#include <hip/hip_runtime.h>
#include <hip/hip_bf16.h>

typedef __attribute__((ext_vector_type(16))) __bf16 v16bf;
typedef __attribute__((ext_vector_type(8)))  float  v8f;
typedef int v4i __attribute__((vector_size(16)));

#define AS1 __attribute__((address_space(1)))
#define AS3 __attribute__((address_space(3)))

union BF16x16 { v16bf v; __bf16 e[16]; unsigned u[8]; };
union F8      { v8f   v; float  f[8]; };

static __device__ __forceinline__ v8f wmma_bf16(v16bf a, v16bf b, v8f c) {
  // D = A(16x32 bf16) * B(32x16 bf16) + C(16x16 f32)
  return __builtin_amdgcn_wmma_f32_16x16x32_bf16(false, a, false, b, (short)0, c,
                                                 false, false);
}

#if __has_builtin(__builtin_amdgcn_global_load_async_to_lds_b128)
#define HAVE_ASYNC_LDS 1
static __device__ __forceinline__ void async_copy_b128(const void* g, void* l) {
  __builtin_amdgcn_global_load_async_to_lds_b128(
      (AS1 v4i*)(char*)(const_cast<void*>(g)), (AS3 v4i*)(char*)l, 0, 0);
}
#else
#define HAVE_ASYNC_LDS 0
#endif

namespace cfg {
constexpr int B = 2, S = 2048, HID = 4096, H = 32, HK = 8, D = 128;
constexpr int NQKV = (H + 2 * HK) * D;  // 6144
constexpr int M = B * S;                // 4096 tokens
}

// ---------------------------------------------------------------------------
// GEMM: C[M,N] = A[M,K] * B[K,N].  A fp32 or bf16 (row-major), B fp32
// (row-major), C fp32.  Block tile 64x128, K-step 32.  8 waves = 2(M) x 4(N),
// each wave computes a 32x32 tile (2x2 WMMA 16x16 tiles).
// B staged COLUMN-major in LDS so B-fragment K-pairs are contiguous dwords.
// ---------------------------------------------------------------------------
template <bool A_BF16>
__global__ __launch_bounds__(256) void gemm_kernel(const void* __restrict__ Ap,
                                                   const float* __restrict__ Bp,
                                                   float* __restrict__ Cp,
                                                   int M, int N, int K) {
  __shared__ __bf16 As[64 * 32];    // row-major [m][k]
  __shared__ __bf16 Bs[128 * 32];   // column-major [n][k]

  const int tid = threadIdx.x;
  const int lane = tid & 31, wave = tid >> 5;
  const int wm = wave >> 2, wn = wave & 3;   // 2 x 4 wave grid
  const int hh = lane >> 4, ll = lane & 15;  // lane-half, lane-in-half
  const int bm = blockIdx.y * 64, bn = blockIdx.x * 128;

  F8 acc[2][2] = {};

  for (int k0 = 0; k0 < K; k0 += 32) {
    __syncthreads();
    // Stage A tile 64x32 (convert to bf16)
    for (int i = tid; i < 64 * 32; i += 256) {
      int r = i >> 5, c = i & 31;
      float x;
      if (A_BF16) x = (float)((const __bf16*)Ap)[(size_t)(bm + r) * K + k0 + c];
      else        x = ((const float*)Ap)[(size_t)(bm + r) * K + k0 + c];
      As[i] = (__bf16)x;
    }
    // Stage B tile 32x128 -> column-major bf16
    for (int i = tid; i < 32 * 128; i += 256) {
      int k = i >> 7, c = i & 127;
      Bs[c * 32 + k] = (__bf16)Bp[(size_t)(k0 + k) * N + bn + c];
    }
    __syncthreads();

    BF16x16 afr[2], bfr[2];
#pragma unroll
    for (int mt = 0; mt < 2; ++mt) {
      const unsigned* arow =
          (const unsigned*)&As[(wm * 32 + mt * 16 + ll) * 32];
#pragma unroll
      for (int v = 0; v < 8; ++v) {
        int du = (v < 4) ? (4 * hh + v) : (8 + 4 * hh + (v - 4));
        afr[mt].u[v] = arow[du];  // K-pair (2*du, 2*du+1)
      }
    }
#pragma unroll
    for (int nt = 0; nt < 2; ++nt) {
      const unsigned* bcol =
          (const unsigned*)&Bs[(wn * 32 + nt * 16 + ll) * 32];
#pragma unroll
      for (int v = 0; v < 8; ++v) bfr[nt].u[v] = bcol[(v < 4) ? (4 * hh + v) : (8 + 4 * hh + (v - 4))];
    }
#pragma unroll
    for (int mt = 0; mt < 2; ++mt)
#pragma unroll
      for (int nt = 0; nt < 2; ++nt)
        acc[mt][nt].v = wmma_bf16(afr[mt].v, bfr[nt].v, acc[mt][nt].v);
  }

  // C/D layout: element (m = r + 8*hh, n = ll) in register r
#pragma unroll
  for (int mt = 0; mt < 2; ++mt)
#pragma unroll
    for (int nt = 0; nt < 2; ++nt) {
      int n = bn + wn * 32 + nt * 16 + ll;
#pragma unroll
      for (int r = 0; r < 8; ++r) {
        int m = bm + wm * 32 + mt * 16 + 8 * hh + r;
        Cp[(size_t)m * N + n] = acc[mt][nt].f[r];
      }
    }
}

// ---------------------------------------------------------------------------
// RoPE (neox) + split/convert QKV fp32 -> bf16 Q[B,H,S,D], K/V[B,HK,S,D]
// ---------------------------------------------------------------------------
__global__ __launch_bounds__(256) void rope_kernel(const int* __restrict__ pos,
                                                   const float* __restrict__ qkv,
                                                   __bf16* __restrict__ Qb,
                                                   __bf16* __restrict__ Kb,
                                                   __bf16* __restrict__ Vb) {
  using namespace cfg;
  const int t = blockIdx.x;  // token index
  const int b = t / S, s = t - b * S;
  const float p = (float)pos[s];
  const float* row = qkv + (size_t)t * NQKV;
  const float kLn = 0.14391157f;  // ln(10000)/64

  for (int idx = threadIdx.x; idx < H * 64; idx += 256) {
    int head = idx >> 6, i = idx & 63;
    float f = p * __expf(-kLn * (float)i);
    float sn, c; __sincosf(f, &sn, &c);
    float x1 = row[head * D + i];
    float x2 = row[head * D + i + 64];
    __bf16* q = Qb + (((size_t)b * H + head) * S + s) * D;
    q[i]      = (__bf16)(x1 * c - x2 * sn);
    q[i + 64] = (__bf16)(x2 * c + x1 * sn);
  }
  for (int idx = threadIdx.x; idx < HK * 64; idx += 256) {
    int head = idx >> 6, i = idx & 63;
    float f = p * __expf(-kLn * (float)i);
    float sn, c; __sincosf(f, &sn, &c);
    float x1 = row[H * D + head * D + i];
    float x2 = row[H * D + head * D + i + 64];
    __bf16* k = Kb + (((size_t)b * HK + head) * S + s) * D;
    k[i]      = (__bf16)(x1 * c - x2 * sn);
    k[i + 64] = (__bf16)(x2 * c + x1 * sn);
  }
  for (int idx = threadIdx.x; idx < HK * D; idx += 256) {
    int head = idx >> 7, d = idx & 127;
    Vb[(((size_t)b * HK + head) * S + s) * D + d] =
        (__bf16)row[(H + HK) * D + idx];
  }
}

// ---------------------------------------------------------------------------
// Flash attention (causal, GQA).  Block = 128 queries for one (b, head);
// 8 waves, wave w owns query rows [q0+16w, q0+16w+16).  Keys in chunks of 64.
// K tiles: double-buffered, filled by GLOBAL_LOAD_ASYNC_TO_LDS_B128 (ASYNCcnt)
// when available.  V tile: staged transposed (column-major) so PV B-fragments
// are contiguous ds_load_b128s.  Online softmax wave-local via shuffles.
// ---------------------------------------------------------------------------
__global__ __launch_bounds__(256) void attn_kernel(const __bf16* __restrict__ Qb,
                                                   const __bf16* __restrict__ Kb,
                                                   const __bf16* __restrict__ Vb,
                                                   __bf16* __restrict__ Ob) {
  using namespace cfg;
  __shared__ __bf16 Ks[2][64 * 128];  // row-major  [key][d], double buffered
  __shared__ __bf16 Vs[128 * 64];     // col-major  [d][key]
  __shared__ __bf16 Ps[8 * 16 * 64];  // per-wave P bounce buffer

  const int tid = threadIdx.x, lane = tid & 31, wave = tid >> 5;
  const int hh = lane >> 4, ll = lane & 15;
  const int q0 = blockIdx.x * 128;
  const int head = blockIdx.y, b = blockIdx.z;
  const int kvh = head >> 2;  // G = 4 query heads per kv head

  const __bf16* Qp = Qb + ((size_t)b * H + head) * S * (size_t)D;
  const __bf16* Kp = Kb + ((size_t)b * HK + kvh) * S * (size_t)D;
  const __bf16* Vp = Vb + ((size_t)b * HK + kvh) * S * (size_t)D;

  // Q A-fragments: row m = ll, K-dim = head dim, 4 fragments of 32
  BF16x16 qf[4];
  {
    const unsigned* qr =
        (const unsigned*)(Qp + (size_t)(q0 + wave * 16 + ll) * D);
#pragma unroll
    for (int dsI = 0; dsI < 4; ++dsI)
#pragma unroll
      for (int v = 0; v < 8; ++v) {
        int ui = (v < 4) ? (dsI * 16 + 4 * hh + v)
                         : (dsI * 16 + 8 + 4 * hh + (v - 4));
        qf[dsI].u[v] = qr[ui];
      }
  }

  F8 oacc[8] = {};
  float run_max[8], run_sum[8];
#pragma unroll
  for (int r = 0; r < 8; ++r) { run_max[r] = -1e30f; run_sum[r] = 0.0f; }

  const float scale = 0.088388347648318447f;  // 1/sqrt(128)
  const int kend = q0 + 128;

#if HAVE_ASYNC_LDS
  // Prologue: async-stage K chunk 0 into buffer 0 (4 b128 ops per wave).
  {
    const char* g = (const char*)Kp;
    char* l = (char*)&Ks[0][0];
#pragma unroll
    for (int n = 0; n < 4; ++n) {
      int off = n * 4096 + tid * 16;
      async_copy_b128(g + off, l + off);
    }
  }
#endif

  for (int k0 = 0; k0 < kend; k0 += 64) {
    const int cur = (k0 >> 6) & 1;
    __syncthreads();  // all waves done reading Vs / Ks[cur^1] from before

    // Stage V chunk (transposed -> column-major)
    {
      const unsigned* vsrc = (const unsigned*)(Vp + (size_t)k0 * D);
      for (int i = tid; i < 64 * 64; i += 256) {
        int key = i >> 6, dp = i & 63;  // d = 2*dp, 2*dp+1
        unsigned pack = vsrc[key * 64 + dp];
        Vs[(2 * dp) * 64 + key]     = ((const __bf16*)&pack)[0];
        Vs[(2 * dp + 1) * 64 + key] = ((const __bf16*)&pack)[1];
      }
    }

#if HAVE_ASYNC_LDS
    if (k0 + 64 < kend) {  // issue next K chunk into other buffer
      const char* g = (const char*)(Kp + (size_t)(k0 + 64) * D);
      char* l = (char*)&Ks[cur ^ 1][0];
#pragma unroll
      for (int n = 0; n < 4; ++n) {
        int off = n * 4096 + tid * 16;
        async_copy_b128(g + off, l + off);
      }
      asm volatile("s_wait_asynccnt 0x4" ::: "memory");  // chunk k0 complete
    } else {
      asm volatile("s_wait_asynccnt 0x0" ::: "memory");
    }
#else
    {  // synchronous K staging fallback
      const unsigned* ksrc = (const unsigned*)(Kp + (size_t)k0 * D);
      unsigned* kdst = (unsigned*)&Ks[cur][0];
      for (int i = tid; i < 64 * 64; i += 256) kdst[i] = ksrc[i];
    }
#endif
    __syncthreads();  // K (async) + V (transposed) visible to all waves

    // ---- S = Q * K^T : per n-tile, preload 4 B-frags then 4 WMMAs ----
    F8 sacc[4] = {};
#pragma unroll
    for (int nt = 0; nt < 4; ++nt) {
      const unsigned* krow =
          (const unsigned*)(&Ks[cur][0] + (nt * 16 + ll) * 128);
      BF16x16 kf[4];  // B-frag: n = key (ll), K = d = dsI*32 + table order (8*hh + 2v, v<4; 16 + 8*hh + 2(v-4), v>=4)
#pragma unroll
      for (int dsI = 0; dsI < 4; ++dsI)
#pragma unroll
        for (int v = 0; v < 8; ++v) kf[dsI].u[v] = krow[dsI * 16 + ((v < 4) ? (4 * hh + v) : (8 + 4 * hh + (v - 4)))];
#pragma unroll
      for (int dsI = 0; dsI < 4; ++dsI)
        sacc[nt].v = wmma_bf16(qf[dsI].v, kf[dsI].v, sacc[nt].v);
    }

    const bool need_mask = (k0 + 63 > q0);
    float pm[4][8];
#pragma unroll
    for (int nt = 0; nt < 4; ++nt)
#pragma unroll
      for (int r = 0; r < 8; ++r) {
        float x = sacc[nt].f[r] * scale;
        if (need_mask) {
          int key = k0 + nt * 16 + ll;
          int qq = q0 + wave * 16 + 8 * hh + r;
          if (key > qq) x = -1e30f;
        }
        pm[nt][r] = x;
      }

    // ---- online softmax (row = register r + lane-half; 16-lane reduce) ----
    float corr[8];
#pragma unroll
    for (int r = 0; r < 8; ++r) {
      float mx = fmaxf(fmaxf(pm[0][r], pm[1][r]), fmaxf(pm[2][r], pm[3][r]));
#pragma unroll
      for (int off = 8; off >= 1; off >>= 1)
        mx = fmaxf(mx, __shfl_xor(mx, off, 32));
      float nm = fmaxf(run_max[r], mx);
      corr[r] = __expf(run_max[r] - nm);
      run_max[r] = nm;
      float rs = 0.0f;
#pragma unroll
      for (int nt = 0; nt < 4; ++nt) {
        pm[nt][r] = __expf(pm[nt][r] - nm);
        rs += pm[nt][r];
      }
#pragma unroll
      for (int off = 8; off >= 1; off >>= 1) rs += __shfl_xor(rs, off, 32);
      run_sum[r] = run_sum[r] * corr[r] + rs;
    }
#pragma unroll
    for (int dt = 0; dt < 8; ++dt)
#pragma unroll
      for (int r = 0; r < 8; ++r) oacc[dt].f[r] *= corr[r];

    // ---- P: C-layout -> A-layout via per-wave LDS bounce ----
    __bf16* pb = Ps + wave * (16 * 64);
#pragma unroll
    for (int nt = 0; nt < 4; ++nt)
#pragma unroll
      for (int r = 0; r < 8; ++r)
        pb[(8 * hh + r) * 64 + nt * 16 + ll] = (__bf16)pm[nt][r];

    asm volatile("s_wait_dscnt 0x0" ::: "memory");  // wave-local LDS RAW fence

    BF16x16 pf[2];
    {
      const unsigned* prow = (const unsigned*)(pb + ll * 64);
#pragma unroll
      for (int ks = 0; ks < 2; ++ks)
#pragma unroll
        for (int v = 0; v < 8; ++v) {
          int ui = (v < 4) ? (ks * 16 + 4 * hh + v)
                           : (ks * 16 + 8 + 4 * hh + (v - 4));
          pf[ks].u[v] = prow[ui];
        }
    }

    // ---- O += P * V : per d-tile, preload 2 B-frags then 2 WMMAs ----
#pragma unroll
    for (int dt = 0; dt < 8; ++dt) {
      const unsigned* vcol = (const unsigned*)(Vs + (dt * 16 + ll) * 64);
      BF16x16 vf[2];  // B-frag: n = d col (ll), K = key = ks*32 + table order (8*hh + 2v, v<4; 16 + 8*hh + 2(v-4), v>=4)
#pragma unroll
      for (int ks = 0; ks < 2; ++ks)
#pragma unroll
        for (int v = 0; v < 8; ++v) vf[ks].u[v] = vcol[ks * 16 + ((v < 4) ? (4 * hh + v) : (8 + 4 * hh + (v - 4)))];
#pragma unroll
      for (int ks = 0; ks < 2; ++ks)
        oacc[dt].v = wmma_bf16(pf[ks].v, vf[ks].v, oacc[dt].v);
    }
  }

  // ---- normalize + store bf16 for O-projection: [token][head*128 + d] ----
#pragma unroll
  for (int dt = 0; dt < 8; ++dt)
#pragma unroll
    for (int r = 0; r < 8; ++r) {
      int qq = q0 + wave * 16 + 8 * hh + r;
      float val = oacc[dt].f[r] / run_sum[r];
      Ob[((size_t)b * S + qq) * (size_t)(H * D) + head * D + dt * 16 + ll] =
          (__bf16)val;
    }
}

// ---------------------------------------------------------------------------
extern "C" void kernel_launch(void* const* d_in, const int* in_sizes, int n_in,
                              void* d_out, int out_size, void* d_ws,
                              size_t ws_size, hipStream_t stream) {
  using namespace cfg;
  const int* positions = (const int*)d_in[0];
  const float* hidden  = (const float*)d_in[1];
  const float* w_qkv   = (const float*)d_in[2];
  const float* w_o     = (const float*)d_in[3];
  float* out = (float*)d_out;

  char* ws = (char*)d_ws;
  float* qkv = (float*)ws;  // [M, NQKV] fp32
  size_t off = (size_t)M * NQKV * sizeof(float);
  __bf16* Qbf = (__bf16*)(ws + off); off += (size_t)B * H  * S * D * sizeof(__bf16);
  __bf16* Kbf = (__bf16*)(ws + off); off += (size_t)B * HK * S * D * sizeof(__bf16);
  __bf16* Vbf = (__bf16*)(ws + off); off += (size_t)B * HK * S * D * sizeof(__bf16);
  __bf16* Abf = (__bf16*)(ws + off);  // [M, H*D] bf16 attention output

  // 1) QKV projection (fp32 -> bf16 WMMA -> fp32)
  gemm_kernel<false><<<dim3(NQKV / 128, M / 64), 256, 0, stream>>>(
      hidden, w_qkv, qkv, M, NQKV, HID);
  // 2) RoPE + split + bf16 conversion
  rope_kernel<<<M, 256, 0, stream>>>(positions, qkv, Qbf, Kbf, Vbf);
  // 3) Causal GQA flash attention
  attn_kernel<<<dim3(S / 128, H, B), 256, 0, stream>>>(Qbf, Kbf, Vbf, Abf);
  // 4) Output projection
  gemm_kernel<true><<<dim3(HID / 128, M / 64), 256, 0, stream>>>(
      Abf, w_o, out, M, HID, HID);
}
